// GATConv_26723286516148
// MI455X (gfx1250) — matrix operand order repaired and checked by reading, not compiled, not run on hardware
//
#include <hip/hip_runtime.h>
#include <hip/hip_bf16.h>

#define NN 50000
#define NE 800000
#define IC 128
#define NH 4
#define OC 32
#define HC 128   // NH*OC

typedef __attribute__((ext_vector_type(16))) __bf16 v16bf;
typedef __attribute__((ext_vector_type(8)))  float  v8f;

// ---------------- zero-init the atomic tables ----------------
__global__ void k_init(float* __restrict__ p, int n) {
    int i = blockIdx.x * blockDim.x + threadIdx.x;
    if (i < n) p[i] = 0.0f;
}

// ---------------- repack W (f32 row-major K x N) into bf16 WMMA B-fragments ----------------
// Fragment f = (ks*8 + nt) holds the 32x16 bf16 B tile for k-block ks, col tile nt,
// laid out so lane L's 16 elements are contiguous at Wp[f*512 + L*16 .. +15].
// Element i of lane L is B[K][N] with K = ks*32 + 16*(i>=8) + 8*(L>=16) + (i&7), N = nt*16 + (L&15).
__global__ void k_packW(const float* __restrict__ W, __bf16* __restrict__ Wp) {
    int o = blockIdx.x * blockDim.x + threadIdx.x;      // 0..16383
    if (o >= 4 * 8 * 32 * 16) return;
    int i    = o & 15;
    int lane = (o >> 4) & 31;
    int nt   = (o >> 9) & 7;
    int ks   = o >> 12;
    int K   = ks * 32 + ((i >> 3) << 4) + ((lane >> 4) << 3) + (i & 7);
    int col = nt * 16 + (lane & 15);
    Wp[o] = (__bf16)W[(size_t)K * HC + col];
}

// ---------------- h = x @ W : one wave -> 16x64 strip (4 WMMA tiles, A reused 4x) ----------
// grid: 3125 blocks x 64 threads (2 waves). wave wv covers col tiles wv*4 .. wv*4+3.
__global__ __launch_bounds__(64) void k_gemm(const float* __restrict__ x,
                                             const __bf16* __restrict__ Wp,
                                             float* __restrict__ h) {
    int mt   = blockIdx.x;             // 0..3124  (16-row strip)
    int wv   = threadIdx.x >> 5;       // 0 | 1
    int lane = threadIdx.x & 31;
    int half = lane >> 4;              // 0 | 1
    int l16  = lane & 15;
    int r0   = mt * 16;
    int ntB  = wv * 4;

    v8f a0 = {}, a1 = {}, a2 = {}, a3 = {};
    #pragma unroll
    for (int ks = 0; ks < 4; ++ks) {
        int kb = ks * 32;
        // A fragment: 16x32 bf16 tile of x (converted on the fly)
        v16bf a;
        const float* xr = x + (size_t)(r0 + l16) * IC + kb + half * 8;
        #pragma unroll
        for (int i = 0; i < 8; ++i) a[i]     = (__bf16)xr[i];        // K = base+0..7
        #pragma unroll
        for (int i = 0; i < 8; ++i) a[8 + i] = (__bf16)xr[16 + i];   // K = base+16..23
        // B fragments: pre-swizzled bf16, one 32B vector per lane per tile
        const v16bf* bp = (const v16bf*)Wp + (size_t)(ks * 8 + ntB) * 32 + lane;
        v16bf b0 = bp[0];
        v16bf b1 = bp[32];
        v16bf b2 = bp[64];
        v16bf b3 = bp[96];
        a0 = __builtin_amdgcn_wmma_f32_16x16x32_bf16(false, a, false, b0, (short)0, a0, false, false);
        a1 = __builtin_amdgcn_wmma_f32_16x16x32_bf16(false, a, false, b1, (short)0, a1, false, false);
        a2 = __builtin_amdgcn_wmma_f32_16x16x32_bf16(false, a, false, b2, (short)0, a2, false, false);
        a3 = __builtin_amdgcn_wmma_f32_16x16x32_bf16(false, a, false, b3, (short)0, a3, false, false);
    }
    // D layout: VGPR v -> row r0 + v + 8*half, col = tile base + l16
    #pragma unroll
    for (int v = 0; v < 8; ++v) {
        float* hr = h + (size_t)(r0 + v + half * 8) * HC + l16;
        hr[(ntB + 0) * 16] = a0[v];
        hr[(ntB + 1) * 16] = a1[v];
        hr[(ntB + 2) * 16] = a2[v];
        hr[(ntB + 3) * 16] = a3[v];
    }
}

// ---------------- s_src / s_dst per (node, head) ----------------
__global__ void k_scores(const float* __restrict__ h, const float* __restrict__ att,
                         float* __restrict__ ssrc, float* __restrict__ sdst) {
    int i = blockIdx.x * blockDim.x + threadIdx.x;
    if (i >= NN * NH) return;
    int n = i >> 2, hd = i & 3;
    const float* hv = h + (size_t)n * HC + hd * OC;
    const float* as = att + hd * 2 * OC;
    const float* ad = as + OC;
    float ss = 0.f, sd = 0.f;
    #pragma unroll
    for (int c = 0; c < OC; ++c) { float v = hv[c]; ss += v * as[c]; sd += v * ad[c]; }
    ssrc[i] = ss; sdst[i] = sd;
}

__device__ __forceinline__ float lrelu(float s) { return s > 0.f ? s : 0.2f * s; }

// ---------------- pass 1: m[row,h] = max(0, max_e alpha) via int-bit atomic max --------------
__global__ void k_edge_max(const long long* __restrict__ ei,
                           const float* __restrict__ ssrc, const float* __restrict__ sdst,
                           int* __restrict__ mbits) {
    int e = blockIdx.x * blockDim.x + threadIdx.x;
    if (e >= NE) return;
    __builtin_prefetch(&ei[e + 8192], 0, 0);
    __builtin_prefetch(&ei[NE + e + 8192], 0, 0);
    int r = (int)ei[e];
    int c = (int)ei[NE + e];
    #pragma unroll
    for (int hd = 0; hd < NH; ++hd) {
        float a  = lrelu(ssrc[r * NH + hd] + sdst[c * NH + hd]);
        float am = a > 0.f ? a : 0.f;                 // all stored values >= 0
        atomicMax(&mbits[r * NH + hd], __float_as_int(am));
    }
}

// ---------------- pass 2: seg_sum of exp(alpha - m[row]) and degree ----------------
__global__ void k_edge_sum(const long long* __restrict__ ei,
                           const float* __restrict__ ssrc, const float* __restrict__ sdst,
                           const float* __restrict__ m,
                           float* __restrict__ ssum, float* __restrict__ deg) {
    int e = blockIdx.x * blockDim.x + threadIdx.x;
    if (e >= NE) return;
    __builtin_prefetch(&ei[e + 8192], 0, 0);
    __builtin_prefetch(&ei[NE + e + 8192], 0, 0);
    int r = (int)ei[e];
    int c = (int)ei[NE + e];
    #pragma unroll
    for (int hd = 0; hd < NH; ++hd) {
        float a = lrelu(ssrc[r * NH + hd] + sdst[c * NH + hd]);
        atomicAdd(&ssum[r * NH + hd], __expf(a - m[r * NH + hd]));
    }
    atomicAdd(&deg[r], 1.0f);
}

// ---------------- denom (in place on ssum) ----------------
__global__ void k_denom(float* __restrict__ ssum, const float* __restrict__ m,
                        const float* __restrict__ deg) {
    int i = blockIdx.x * blockDim.x + threadIdx.x;
    if (i >= NN * NH) return;
    int n = i >> 2;
    ssum[i] = ssum[i] + ((float)NE - deg[n]) * __expf(-m[i]);
}

// ---------------- pass 3: coef[col,h] += exp(alpha - m[row]) / denom[row,h] ----------------
__global__ void k_edge_coef(const long long* __restrict__ ei,
                            const float* __restrict__ ssrc, const float* __restrict__ sdst,
                            const float* __restrict__ m, const float* __restrict__ denom,
                            float* __restrict__ coef) {
    int e = blockIdx.x * blockDim.x + threadIdx.x;
    if (e >= NE) return;
    __builtin_prefetch(&ei[e + 8192], 0, 0);
    __builtin_prefetch(&ei[NE + e + 8192], 0, 0);
    int r = (int)ei[e];
    int c = (int)ei[NE + e];
    #pragma unroll
    for (int hd = 0; hd < NH; ++hd) {
        float a  = lrelu(ssrc[r * NH + hd] + sdst[c * NH + hd]);
        float an = __expf(a - m[r * NH + hd]) / denom[r * NH + hd];
        atomicAdd(&coef[c * NH + hd], an);
    }
}

// ---------------- out = h * coef(broadcast per head) + bias ----------------
__global__ void k_out(const float* __restrict__ h, const float* __restrict__ coef,
                      const float* __restrict__ bias, float* __restrict__ out) {
    int i = blockIdx.x * blockDim.x + threadIdx.x;   // over NN*32 float4s
    if (i >= NN * 32) return;
    int n = i >> 5, q = i & 31;
    int hd = q >> 3;                                  // 8 float4s per head
    float k = coef[n * NH + hd];
    const float4 hv = ((const float4*)h)[(size_t)n * 32 + q];
    const float4 bv = ((const float4*)bias)[q];
    float4 o;
    o.x = hv.x * k + bv.x; o.y = hv.y * k + bv.y;
    o.z = hv.z * k + bv.z; o.w = hv.w * k + bv.w;
    ((float4*)out)[(size_t)n * 32 + q] = o;
}

extern "C" void kernel_launch(void* const* d_in, const int* in_sizes, int n_in,
                              void* d_out, int out_size, void* d_ws, size_t ws_size,
                              hipStream_t stream) {
    const float*     x    = (const float*)d_in[0];
    const long long* ei   = (const long long*)d_in[1];   // int64 edge_index [2, NE]
    const float*     W    = (const float*)d_in[2];
    const float*     att  = (const float*)d_in[3];
    const float*     bias = (const float*)d_in[4];
    float*           out  = (float*)d_out;

    // workspace layout (floats)
    float* hbuf = (float*)d_ws;              // NN*128
    float* mbuf = hbuf + (size_t)NN * HC;    // NN*4   (max table, nonneg float bits)
    float* ssum = mbuf + NN * NH;            // NN*4   (later: denom)
    float* coef = ssum + NN * NH;            // NN*4
    float* deg  = coef + NN * NH;            // NN
    float* ssrc = deg  + NN;                 // NN*4
    float* sdst = ssrc + NN * NH;            // NN*4
    __bf16* Wp  = (__bf16*)(sdst + NN * NH); // 16384 bf16 (32 KB), 32B-aligned

    const int zeroN = NN * (NH * 3 + 1);     // mbuf..deg contiguous
    k_init<<<(zeroN + 255) / 256, 256, 0, stream>>>(mbuf, zeroN);

    k_packW<<<64, 256, 0, stream>>>(W, Wp);
    k_gemm<<<3125, 64, 0, stream>>>(x, Wp, hbuf);       // 6250 waves, 16 WMMA each

    k_scores<<<(NN * NH + 255) / 256, 256, 0, stream>>>(hbuf, att, ssrc, sdst);

    const int eb = (NE + 255) / 256;
    k_edge_max<<<eb, 256, 0, stream>>>(ei, ssrc, sdst, (int*)mbuf);
    k_edge_sum<<<eb, 256, 0, stream>>>(ei, ssrc, sdst, mbuf, ssum, deg);
    k_denom<<<(NN * NH + 255) / 256, 256, 0, stream>>>(ssum, mbuf, deg);
    k_edge_coef<<<eb, 256, 0, stream>>>(ei, ssrc, sdst, mbuf, ssum, coef);
    k_out<<<(NN * 32 + 255) / 256, 256, 0, stream>>>(hbuf, coef, bias, out);
}
